// GraphReader_2336462209715
// MI455X (gfx1250) — compile-verified, matrix operand order checked by reading, not run on hardware
//
#include <hip/hip_runtime.h>
#include <cstddef>

// Problem constants (match reference)
constexpr int H  = 768;
constexpr int NH = 8;
constexpr int DH = 96;
constexpr int B  = 64;
constexpr int N  = 131072;
constexpr float SCALE = 0.1020620726159657f; // 1/sqrt(96)

typedef _Float16 v16h __attribute__((ext_vector_type(16)));
typedef _Float16 v8h  __attribute__((ext_vector_type(8)));
typedef float    v8f  __attribute__((ext_vector_type(8)));
typedef float    v4f  __attribute__((ext_vector_type(4)));

// ---------------------------------------------------------------------------
// Zero the y accumulator (atomically accumulated each launch).
// ---------------------------------------------------------------------------
__global__ __launch_bounds__(256) void k_zero(float* __restrict__ p, int n) {
  int i = blockIdx.x * 256 + threadIdx.x;
  if (i < n) p[i] = 0.f;
}

// ---------------------------------------------------------------------------
// Per-graph query pipeline: qp = query@Wp.T+bp ; q = (qp@Wq.T+bq)*scale ;
// qk[b,h,c] = sum_d q[h,d]*Wk[h*96+d,c] ; sb[b,h] = q[h,:]·bk[h,:]
// qk_h is padded to 16 "head" rows per graph; rows 8..15 are zero so the
// WMMA B-matrix padding columns need no conditional loads.
// One block per graph b.
// ---------------------------------------------------------------------------
__global__ __launch_bounds__(256) void k_prep(
    const float* __restrict__ query,
    const float* __restrict__ Wp, const float* __restrict__ bp,
    const float* __restrict__ Wq, const float* __restrict__ bq,
    const float* __restrict__ Wk, const float* __restrict__ bk,
    float* __restrict__ qk_f, _Float16* __restrict__ qk_h,
    float* __restrict__ sb) {
  int b = blockIdx.x, tid = threadIdx.x;
  __shared__ float qin[H], qp[H], qv[H];
  for (int c = tid; c < H; c += 256) qin[c] = query[(size_t)b * H + c];
  __syncthreads();
  for (int c = tid; c < H; c += 256) {
    const float* wr = Wp + (size_t)c * H;
    float s = bp[c];
    for (int k = 0; k < H; ++k) s += qin[k] * wr[k];
    qp[c] = s;
  }
  __syncthreads();
  for (int c = tid; c < H; c += 256) {
    const float* wr = Wq + (size_t)c * H;
    float s = bq[c];
    for (int k = 0; k < H; ++k) s += qp[k] * wr[k];
    qv[c] = s * SCALE;
  }
  __syncthreads();
  for (int idx = tid; idx < NH * H; idx += 256) {
    int h = idx / H, c = idx - h * H;
    const float* q_ = qv + h * DH;
    float s = 0.f;
    for (int d = 0; d < DH; ++d) s += q_[d] * Wk[((size_t)(h * DH + d)) * H + c];
    qk_f[((size_t)b * NH + h) * H + c] = s;
    qk_h[((size_t)b * 16 + h) * H + c] = (_Float16)s;
  }
  // zero the 8 padding rows of qk_h
  for (int idx = tid; idx < NH * H; idx += 256) {
    int h = idx / H, c = idx - h * H;
    qk_h[((size_t)b * 16 + 8 + h) * H + c] = (_Float16)0.f;
  }
  if (tid < NH) {
    float s = 0.f;
    for (int d = 0; d < DH; ++d) s += qv[tid * DH + d] * bk[tid * DH + d];
    sb[b * NH + tid] = s;
  }
}

// ---------------------------------------------------------------------------
// Segment boundaries: bounds[b] = first index with batch_indices[i] >= b.
// ---------------------------------------------------------------------------
__global__ void k_bounds(const int* __restrict__ bidx, int* __restrict__ bounds) {
  int t = threadIdx.x;
  if (t <= B) {
    int lo = 0, hi = N;
    while (lo < hi) {
      int mid = (lo + hi) >> 1;
      if (bidx[mid] < t) lo = mid + 1; else hi = mid;
    }
    bounds[t] = lo;
  }
}

// ---------------------------------------------------------------------------
// Pass 1 (streaming): s[n,h] = x[n,:]·qk[b(n),h,:] + sb[b(n),h]
// WMMA f16 16x16x32 with f32 accumulate. Per §7.12.2 the 16-bit fragment
// swizzle for lane group g covers two contiguous runs [kk+8g, +8) and
// [kk+16+8g, +8): A = 4x b128 loads (f32), B = 2x b128 loads (f16),
// completely branch-free (qk_h padding rows are pre-zeroed).
// Wave-uniform scalar fallback for graph-boundary tiles.
// ---------------------------------------------------------------------------
__global__ __launch_bounds__(256) void k_scores(
    const float* __restrict__ x, const int* __restrict__ bidx,
    const _Float16* __restrict__ qk_h, const float* __restrict__ qk_f,
    const float* __restrict__ sb, float* __restrict__ s_out) {
  int wave = threadIdx.x >> 5;
  int lane = threadIdx.x & 31;
  int sub  = lane & 15;
  int grp  = lane >> 4;
  for (int t = 0; t < 2; ++t) {
    int tile  = blockIdx.x * 16 + wave * 2 + t;
    int node0 = tile * 16;
    int b0 = bidx[node0];
    bool uni = __all((int)(bidx[node0 + sub] == b0)) != 0;
    if (uni) {
      const float*    xrow = x + (size_t)(node0 + sub) * H + 8 * grp;
      const _Float16* qrow = qk_h + ((size_t)b0 * 16 + sub) * H + 8 * grp;
      v8f acc = {};
      #pragma unroll 2
      for (int kk = 0; kk < H; kk += 32) {
        __builtin_prefetch(xrow + kk + 256, 0, 1); // global_prefetch_b8
        v4f a0 = *(const v4f*)(xrow + kk);
        v4f a1 = *(const v4f*)(xrow + kk + 4);
        v4f a2 = *(const v4f*)(xrow + kk + 16);
        v4f a3 = *(const v4f*)(xrow + kk + 20);
        v8h b0v = *(const v8h*)(qrow + kk);
        v8h b1v = *(const v8h*)(qrow + kk + 16);
        v16h a, bf;
        #pragma unroll
        for (int e = 0; e < 4; ++e) {
          a[e]      = (_Float16)a0[e];
          a[e + 4]  = (_Float16)a1[e];
          a[e + 8]  = (_Float16)a2[e];
          a[e + 12] = (_Float16)a3[e];
        }
        #pragma unroll
        for (int e = 0; e < 8; ++e) {
          bf[e]     = b0v[e];
          bf[e + 8] = b1v[e];
        }
        acc = __builtin_amdgcn_wmma_f32_16x16x32_f16(
            false, a, false, bf, (short)0, acc, false, false);
      }
      if (sub < NH) {
        float sv = sb[b0 * NH + sub];
        #pragma unroll
        for (int r = 0; r < 8; ++r) {
          int m = r + 8 * grp;                   // D layout: VGPR r, lane group
          s_out[(size_t)(node0 + m) * NH + sub] = acc[r] + sv;
        }
      }
    } else {
      if (lane < 16) {                           // scalar fp32 fallback
        int node = node0 + lane;
        int b = bidx[node];
        const float* xr = x + (size_t)node * H;
        for (int h = 0; h < NH; ++h) {
          const float* qr = qk_f + ((size_t)b * NH + h) * H;
          float sm = sb[b * NH + h];
          for (int c = 0; c < H; ++c) sm += xr[c] * qr[c];
          s_out[(size_t)node * NH + h] = sm;
        }
      }
    }
  }
}

// ---------------------------------------------------------------------------
// Segment max + sum(exp). One block per graph.
// ---------------------------------------------------------------------------
__global__ __launch_bounds__(256) void k_segred(
    const float* __restrict__ s, const int* __restrict__ bounds,
    float* __restrict__ mb, float* __restrict__ db) {
  int b = blockIdx.x, tid = threadIdx.x;
  int start = bounds[b], end = bounds[b + 1];
  int h = tid & 7, slot = tid >> 3;
  __shared__ float red[256];
  float v = -3.402823e38f;
  for (int n = start + slot; n < end; n += 32) v = fmaxf(v, s[(size_t)n * NH + h]);
  red[tid] = v; __syncthreads();
  for (int off = 16; off >= 1; off >>= 1) {
    if (slot < off) red[tid] = fmaxf(red[tid], red[tid + off * 8]);
    __syncthreads();
  }
  float m = red[h];
  __syncthreads();
  float sum = 0.f;
  for (int n = start + slot; n < end; n += 32) sum += expf(s[(size_t)n * NH + h] - m);
  red[tid] = sum; __syncthreads();
  for (int off = 16; off >= 1; off >>= 1) {
    if (slot < off) red[tid] += red[tid + off * 8];
    __syncthreads();
  }
  if (tid < NH) { mb[b * NH + h] = m; db[b * NH + h] = red[h]; }
}

// ---------------------------------------------------------------------------
// Pass 2 (streaming): y[b,h,c] += w[n,h]*x[n,c] with register-resident
// partials; flush via fp32 atomics when the graph id changes (sorted input).
// Each thread owns columns {tid, tid+256, tid+512} for all 8 heads.
// ---------------------------------------------------------------------------
__global__ __launch_bounds__(256) void k_accum(
    const float* __restrict__ x, const int* __restrict__ bidx,
    const float* __restrict__ s, const float* __restrict__ mb,
    const float* __restrict__ db, float* __restrict__ y) {
  int base = blockIdx.x * 256, tid = threadIdx.x;
  float acc[24];
  #pragma unroll
  for (int i = 0; i < 24; ++i) acc[i] = 0.f;
  int cur_b = bidx[base];
  __shared__ float wL[256];
  __shared__ int   bL[32];
  auto flush = [&]() {
    #pragma unroll
    for (int hh = 0; hh < 8; ++hh)
      #pragma unroll
      for (int j = 0; j < 3; ++j) {
        atomicAdd(&y[((size_t)cur_b * NH + hh) * H + tid + 256 * j], acc[hh * 3 + j]);
        acc[hh * 3 + j] = 0.f;
      }
  };
  for (int g = 0; g < 8; ++g) {
    int i32 = tid >> 3, h = tid & 7;
    int n = base + g * 32 + i32;
    int bb = bidx[n];
    float d   = db[bb * NH + h];
    float inv = d > 0.f ? 1.f / d : 0.f;
    wL[tid] = expf(s[(size_t)n * NH + h] - mb[bb * NH + h]) * inv;
    if (h == 0) bL[i32] = bb;
    __syncthreads();
    for (int i = 0; i < 32; ++i) {
      int nb = bL[i];
      if (nb != cur_b) { flush(); cur_b = nb; }   // block-uniform
      const float* xr = x + (size_t)(base + g * 32 + i) * H;
      __builtin_prefetch(xr + H + tid, 0, 1);
      float x0 = xr[tid], x1 = xr[tid + 256], x2 = xr[tid + 512];
      #pragma unroll
      for (int hh = 0; hh < 8; ++hh) {
        float wv = wL[i * 8 + hh];
        acc[hh * 3 + 0] += wv * x0;
        acc[hh * 3 + 1] += wv * x1;
        acc[hh * 3 + 2] += wv * x2;
      }
    }
    __syncthreads();
  }
  flush();
}

// ---------------------------------------------------------------------------
// ctx = y @ Wv_head.T + bv*1[nonempty]; attn = ctx @ Wo.T + bo (masked);
// LayerNorm. One block per graph.
// ---------------------------------------------------------------------------
__global__ __launch_bounds__(256) void k_final(
    const float* __restrict__ y,
    const float* __restrict__ Wv, const float* __restrict__ bv,
    const float* __restrict__ Wo, const float* __restrict__ bo,
    const float* __restrict__ gamma, const float* __restrict__ beta,
    const float* __restrict__ db, float* __restrict__ out) {
  int b = blockIdx.x, tid = threadIdx.x;
  __shared__ float yL[NH * H];
  __shared__ float ctxL[H];
  __shared__ float aoL[H];
  __shared__ float red[256];
  for (int i = tid; i < NH * H; i += 256) yL[i] = y[(size_t)b * NH * H + i];
  bool nonempty = db[b * NH] > 0.f;
  __syncthreads();
  for (int idx = tid; idx < H; idx += 256) {        // idx = h*96 + d
    int h = idx / DH;
    const float* wr = Wv + (size_t)idx * H;
    const float* yh = yL + h * H;
    float s = 0.f;
    for (int c = 0; c < H; ++c) s += yh[c] * wr[c];
    ctxL[idx] = s + bv[idx];                         // sum_n w = 1 if nonempty
  }
  __syncthreads();
  for (int e = tid; e < H; e += 256) {
    const float* wr = Wo + (size_t)e * H;
    float s = bo[e];
    for (int f = 0; f < H; ++f) s += ctxL[f] * wr[f];
    aoL[e] = nonempty ? s : 0.f;
  }
  __syncthreads();
  float ps = 0.f, pq = 0.f;
  for (int e = tid; e < H; e += 256) { float v = aoL[e]; ps += v; pq += v * v; }
  red[tid] = ps; __syncthreads();
  for (int off = 128; off >= 1; off >>= 1) { if (tid < off) red[tid] += red[tid + off]; __syncthreads(); }
  float mu = red[0] / (float)H;
  __syncthreads();
  red[tid] = pq; __syncthreads();
  for (int off = 128; off >= 1; off >>= 1) { if (tid < off) red[tid] += red[tid + off]; __syncthreads(); }
  float var = red[0] / (float)H - mu * mu;
  if (var < 0.f) var = 0.f;
  float rs = rsqrtf(var + 1e-5f);
  for (int e = tid; e < H; e += 256)
    out[(size_t)b * H + e] = (aoL[e] - mu) * rs * gamma[e] + beta[e];
}

// ---------------------------------------------------------------------------
extern "C" void kernel_launch(void* const* d_in, const int* in_sizes, int n_in,
                              void* d_out, int out_size, void* d_ws, size_t ws_size,
                              hipStream_t stream) {
  const float* query = (const float*)d_in[0];
  const float* x     = (const float*)d_in[1];
  const int*   bidx  = (const int*)d_in[2];
  const float* Wp = (const float*)d_in[3];
  const float* bp = (const float*)d_in[4];
  const float* Wq = (const float*)d_in[5];
  const float* bq = (const float*)d_in[6];
  const float* Wk = (const float*)d_in[7];
  const float* bk = (const float*)d_in[8];
  const float* Wv = (const float*)d_in[9];
  const float* bv = (const float*)d_in[10];
  const float* Wo = (const float*)d_in[11];
  const float* bo = (const float*)d_in[12];
  const float* gamma = (const float*)d_in[13];
  const float* beta  = (const float*)d_in[14];

  char* ws = (char*)d_ws;
  size_t o = 0;
  float*    qk_f = (float*)(ws + o);    o += (size_t)B * NH * H * 4;  // 1.5 MB
  _Float16* qk_h = (_Float16*)(ws + o); o += (size_t)B * 16 * H * 2;  // 1.5 MB (padded)
  float*    sb   = (float*)(ws + o);    o += 4096;
  float*    sbuf = (float*)(ws + o);    o += (size_t)N * NH * 4;      // 4 MB
  int*      bnds = (int*)(ws + o);      o += 1024;
  float*    mb   = (float*)(ws + o);    o += 4096;
  float*    db   = (float*)(ws + o);    o += 4096;
  float*    y    = (float*)(ws + o);    o += (size_t)B * NH * H * 4;  // 1.5 MB

  k_zero  <<<(B * NH * H + 255) / 256, 256, 0, stream>>>(y, B * NH * H);
  k_prep  <<<B, 256, 0, stream>>>(query, Wp, bp, Wq, bq, Wk, bk, qk_f, qk_h, sb);
  k_bounds<<<1, 96, 0, stream>>>(bidx, bnds);
  k_scores<<<N / 256, 256, 0, stream>>>(x, bidx, qk_h, qk_f, sb, sbuf);
  k_segred<<<B, 256, 0, stream>>>(sbuf, bnds, mb, db);
  k_accum <<<N / 256, 256, 0, stream>>>(x, bidx, sbuf, mb, db, y);
  k_final <<<B, 256, 0, stream>>>(y, Wv, bv, Wo, bo, gamma, beta, db, (float*)d_out);
}
